// DepthVolume2D_6468220748331
// MI455X (gfx1250) — compile-verified, matrix operand order checked by reading, not run on hardware
//
#include <hip/hip_runtime.h>
#include <hip/hip_bf16.h>
#include <math.h>

// ---------------------------------------------------------------------------
// CDNA5 (gfx1250) DepthVolume2D. All 3x3 convs are implicit GEMMs on
// V_WMMA_F32_16X16X32_F16: pre-packed A fragments, f16 halo/channel-padded
// inputs (pow2 channels -> shift/mask K decode), LDS-staged offset table,
// two 16-pixel output tiles per wave (2 wmma per chunk sharing A + offsets).
// ---------------------------------------------------------------------------

typedef __attribute__((ext_vector_type(16))) _Float16 v16h;
typedef __attribute__((ext_vector_type(8)))  float    v8f;

#define DEV __device__ __forceinline__

DEV int imin_(int a, int b) { return a < b ? a : b; }

// ---------------- weight packer: A-fragment layout (ISA 7.12.2) -------------
// wp[chunk*512 + lane*16 + e] ; k = chunk*32 + ((e>>3)<<4) + ((lane>>4)<<3) + (e&7)
__global__ void k_packw(const float* __restrict__ w, _Float16* __restrict__ wp,
                        int Cin, int Cout, int Cmask, int shift, int Kround) {
  int i = blockIdx.x * blockDim.x + threadIdx.x;
  int n = (Kround >> 5) * 512;
  if (i >= n) return;
  int chunk = i >> 9, r = i & 511;
  int lane = r >> 4, e = r & 15;
  int k = (chunk << 5) + ((e >> 3) << 4) + ((lane >> 4) << 3) + (e & 7);
  int m = lane & 15;
  int ci = k & Cmask, tap = k >> shift;
  float v = 0.f;
  if (tap < 9 && ci < Cin && m < Cout) v = w[(m * Cin + ci) * 9 + tap];
  wp[i] = (_Float16)v;
}

// ---------------- per-layer gather-offset table -----------------------------
__global__ void k_offtab(int* __restrict__ off, int Cmask, int shift,
                         int Hp, int Wp, int dil, int Kround) {
  int i = blockIdx.x * blockDim.x + threadIdx.x;
  if (i >= Kround) return;
  int ci = i & Cmask;
  int tap = i >> shift; if (tap > 8) tap = 8;   // overshoot -> A is zero anyway
  int ky = tap / 3, kx = tap - 3 * ky;
  off[i] = (ci * Hp + ky * dil) * Wp + kx * dil;
}

// ---------------- f32 -> f16 halo-padded, channel-padded input --------------
__global__ void k_padcvt(const float* __restrict__ in, _Float16* __restrict__ out,
                         int B, int Cin, int Cpad, int H, int W, int dil) {
  int Hp = H + 2 * dil, Wp = W + 2 * dil;
  int i = blockIdx.x * blockDim.x + threadIdx.x;
  int n = B * Cpad * Hp * Wp;
  if (i >= n) return;
  int b = i / (Cpad * Hp * Wp), r = i - b * Cpad * Hp * Wp;
  int c = r / (Hp * Wp); r -= c * (Hp * Wp);
  int y = r / Wp, x = r - y * Wp;
  float v = 0.f;
  int yy = y - dil, xx = x - dil;
  if (c < Cin && yy >= 0 && yy < H && xx >= 0 && xx < W)
    v = in[((b * Cin + c) * H + yy) * W + xx];
  out[i] = (_Float16)v;
}

// ----------------------------- WMMA conv ------------------------------------
// act: 0 = +bias  1 = relu(bn(+bias))  2 = sigmoid(+bias)  3 = tanh(+bias)
// One wave computes a 16(Cout) x 32(pixels) output block = 2 WMMA tiles that
// share the A fragment and the LDS-resident gather offsets.
__global__ void conv3x3_wmma3(const _Float16* __restrict__ inh,
                              const _Float16* __restrict__ wpack,
                              const int* __restrict__ offtab,
                              int Kround, int nchunk,
                              const float* __restrict__ bias,
                              const float* __restrict__ gamma,
                              const float* __restrict__ beta,
                              float* __restrict__ out,
                              int B, int Cpad, int Cout, int H, int W,
                              int Hp, int Wp, int act) {
  __shared__ int soff[320];
  for (int i = threadIdx.x; i < Kround; i += blockDim.x) soff[i] = offtab[i];
  __syncthreads();

  const int lane = threadIdx.x & 31;
  const int wave = threadIdx.x >> 5;
  const int wpb  = blockDim.x >> 5;
  const int tile = blockIdx.x * wpb + wave;
  const int HW = H * W;
  const int tilesPerImg = (HW + 31) >> 5;
  if (tile >= B * tilesPerImg) return;            // wave-uniform exit
  const int b  = tile / tilesPerImg;
  const int p0 = (tile - b * tilesPerImg) << 5;
  const int col = lane & 15;
  int pixA = p0 + col;
  int pixB = p0 + 16 + col;
  const bool vA = pixA < HW; if (!vA) pixA = HW - 1;
  const bool vB = pixB < HW; if (!vB) pixB = HW - 1;
  const int pyA = pixA / W, pxA = pixA - pyA * W;
  const int pyB = pixB / W, pxB = pixB - pyB * W;
  const _Float16* ib = inh + (size_t)b * Cpad * Hp * Wp;
  const _Float16* pA = ib + (pyA * Wp + pxA);     // per-lane base pointers
  const _Float16* pB = ib + (pyB * Wp + pxB);
  const int kHalf = (lane >> 4) << 3;             // 0 or 8

  v8f accA = {}, accB = {};
#pragma unroll 2
  for (int c = 0; c < nchunk; ++c) {
    v16h afrag = *reinterpret_cast<const v16h*>(wpack + (size_t)(((c << 5) + lane) << 4));
    const int base = (c << 5) + kHalf;
    int4 o0 = *reinterpret_cast<const int4*>(&soff[base]);
    int4 o1 = *reinterpret_cast<const int4*>(&soff[base + 4]);
    int4 o2 = *reinterpret_cast<const int4*>(&soff[base + 16]);
    int4 o3 = *reinterpret_cast<const int4*>(&soff[base + 20]);
    const unsigned o[16] = {
      (unsigned)o0.x, (unsigned)o0.y, (unsigned)o0.z, (unsigned)o0.w,
      (unsigned)o1.x, (unsigned)o1.y, (unsigned)o1.z, (unsigned)o1.w,
      (unsigned)o2.x, (unsigned)o2.y, (unsigned)o2.z, (unsigned)o2.w,
      (unsigned)o3.x, (unsigned)o3.y, (unsigned)o3.z, (unsigned)o3.w };
    v16h fa, fb;
#pragma unroll
    for (int e = 0; e < 16; ++e) fa[e] = pA[(size_t)o[e]];
#pragma unroll
    for (int e = 0; e < 16; ++e) fb[e] = pB[(size_t)o[e]];
    accA = __builtin_amdgcn_wmma_f32_16x16x32_f16(
        false, afrag, false, fa, (short)0, accA, false, false);
    accB = __builtin_amdgcn_wmma_f32_16x16x32_f16(
        false, afrag, false, fb, (short)0, accB, false, false);
  }

  const float bnscale = 0.99999500003749979f;     // rsqrt(1 + 1e-5)
#pragma unroll
  for (int r = 0; r < 8; ++r) {
    int M = r + ((lane >> 4) << 3);               // D layout (ISA 7.12.2)
    if (M >= Cout) continue;
    float s  = bias[M];
    float g_ = (act == 1) ? gamma[M] * bnscale : 0.f;
    float b_ = (act == 1) ? beta[M] : 0.f;
    float va = accA[r] + s;
    float vb = accB[r] + s;
    if (act == 1)      { va = fmaxf(va * g_ + b_, 0.f);  vb = fmaxf(vb * g_ + b_, 0.f); }
    else if (act == 2) { va = 1.f / (1.f + __expf(-va)); vb = 1.f / (1.f + __expf(-vb)); }
    else if (act == 3) { va = tanhf(va);                 vb = tanhf(vb); }
    if (vA) out[((b * Cout + M) * H + pyA) * W + pxA] = va;
    if (vB) out[((b * Cout + M) * H + pyB) * W + pxB] = vb;
  }
}

// --------------------------- small helpers ----------------------------------
__global__ void k_downsample(const float* __restrict__ src, float* __restrict__ dst) {
  int i = blockIdx.x * blockDim.x + threadIdx.x;
  if (i >= 9 * 128 * 128) return;
  int cc = i / (128 * 128);
  int r  = i - cc * 128 * 128;
  int y  = r / 128, x = r - y * 128;
  dst[i] = src[(cc * 256 + 2 * y) * 256 + 2 * x];
}

DEV void inv4_(const float m[16], float o[16]) {
  float inv[16];
  inv[0]= m[5]*m[10]*m[15]-m[5]*m[11]*m[14]-m[9]*m[6]*m[15]+m[9]*m[7]*m[14]+m[13]*m[6]*m[11]-m[13]*m[7]*m[10];
  inv[4]=-m[4]*m[10]*m[15]+m[4]*m[11]*m[14]+m[8]*m[6]*m[15]-m[8]*m[7]*m[14]-m[12]*m[6]*m[11]+m[12]*m[7]*m[10];
  inv[8]= m[4]*m[9]*m[15]-m[4]*m[11]*m[13]-m[8]*m[5]*m[15]+m[8]*m[7]*m[13]+m[12]*m[5]*m[11]-m[12]*m[7]*m[9];
  inv[12]=-m[4]*m[9]*m[14]+m[4]*m[10]*m[13]+m[8]*m[5]*m[14]-m[8]*m[6]*m[13]-m[12]*m[5]*m[10]+m[12]*m[6]*m[9];
  inv[1]=-m[1]*m[10]*m[15]+m[1]*m[11]*m[14]+m[9]*m[2]*m[15]-m[9]*m[3]*m[14]-m[13]*m[2]*m[11]+m[13]*m[3]*m[10];
  inv[5]= m[0]*m[10]*m[15]-m[0]*m[11]*m[14]-m[8]*m[2]*m[15]+m[8]*m[3]*m[14]+m[12]*m[2]*m[11]-m[12]*m[3]*m[10];
  inv[9]=-m[0]*m[9]*m[15]+m[0]*m[11]*m[13]+m[8]*m[1]*m[15]-m[8]*m[3]*m[13]-m[12]*m[1]*m[11]+m[12]*m[3]*m[9];
  inv[13]= m[0]*m[9]*m[14]-m[0]*m[10]*m[13]-m[8]*m[1]*m[14]+m[8]*m[2]*m[13]+m[12]*m[1]*m[10]-m[12]*m[2]*m[9];
  inv[2]= m[1]*m[6]*m[15]-m[1]*m[7]*m[14]-m[5]*m[2]*m[15]+m[5]*m[3]*m[14]+m[13]*m[2]*m[7]-m[13]*m[3]*m[6];
  inv[6]=-m[0]*m[6]*m[15]+m[0]*m[7]*m[14]+m[4]*m[2]*m[15]-m[4]*m[3]*m[14]-m[12]*m[2]*m[7]+m[12]*m[3]*m[6];
  inv[10]= m[0]*m[5]*m[15]-m[0]*m[7]*m[13]-m[4]*m[1]*m[15]+m[4]*m[3]*m[13]+m[12]*m[1]*m[7]-m[12]*m[3]*m[5];
  inv[14]=-m[0]*m[5]*m[14]+m[0]*m[6]*m[13]+m[4]*m[1]*m[14]-m[4]*m[2]*m[13]-m[12]*m[1]*m[6]+m[12]*m[2]*m[5];
  inv[3]=-m[1]*m[6]*m[11]+m[1]*m[7]*m[10]+m[5]*m[2]*m[11]-m[5]*m[3]*m[10]-m[9]*m[2]*m[7]+m[9]*m[3]*m[6];
  inv[7]= m[0]*m[6]*m[11]-m[0]*m[7]*m[10]-m[4]*m[2]*m[11]+m[4]*m[3]*m[10]+m[8]*m[2]*m[7]-m[8]*m[3]*m[6];
  inv[11]=-m[0]*m[5]*m[11]+m[0]*m[7]*m[9]+m[4]*m[1]*m[11]-m[4]*m[3]*m[9]-m[8]*m[1]*m[7]+m[8]*m[3]*m[5];
  inv[15]= m[0]*m[5]*m[10]-m[0]*m[6]*m[9]-m[4]*m[1]*m[10]+m[4]*m[2]*m[9]+m[8]*m[1]*m[6]-m[8]*m[2]*m[5];
  float det = m[0]*inv[0]+m[1]*inv[4]+m[2]*inv[8]+m[3]*inv[12];
  det = 1.0f / det;
  for (int i = 0; i < 16; ++i) o[i] = inv[i] * det;
}

DEV void inv3_(const float a[9], float o[9]) {
  float d = a[0]*(a[4]*a[8]-a[5]*a[7]) - a[1]*(a[3]*a[8]-a[5]*a[6]) + a[2]*(a[3]*a[7]-a[4]*a[6]);
  float id = 1.0f / d;
  o[0]=(a[4]*a[8]-a[5]*a[7])*id; o[1]=(a[2]*a[7]-a[1]*a[8])*id; o[2]=(a[1]*a[5]-a[2]*a[4])*id;
  o[3]=(a[5]*a[6]-a[3]*a[8])*id; o[4]=(a[0]*a[8]-a[2]*a[6])*id; o[5]=(a[2]*a[3]-a[0]*a[5])*id;
  o[6]=(a[3]*a[7]-a[4]*a[6])*id; o[7]=(a[1]*a[6]-a[0]*a[7])*id; o[8]=(a[0]*a[4]-a[1]*a[3])*id;
}

__global__ void k_setup_geom(const float* __restrict__ dst_intr,
                             const float* __restrict__ dst_extr,
                             const float* __restrict__ src_intr,
                             const float* __restrict__ src_extr,
                             float* __restrict__ geom) {
  if (threadIdx.x != 0 || blockIdx.x != 0) return;
  float Kd[9];
  for (int i = 0; i < 9; ++i) Kd[i] = dst_intr[i];
  Kd[0]*=0.5f; Kd[2]*=0.5f; Kd[4]*=0.5f; Kd[5]*=0.5f;
  float Edinv[16]; inv4_(dst_extr, Edinv);
  float Kdi[9];    inv3_(Kd, Kdi);
  for (int v = 0; v < 3; ++v) {
    const float* E = src_extr + v * 16;
    float T[16];
    for (int r = 0; r < 4; ++r)
      for (int c = 0; c < 4; ++c) {
        float s = 0.f;
        for (int k = 0; k < 4; ++k) s += E[r*4+k] * Edinv[k*4+c];
        T[r*4+c] = s;
      }
    float Ks[9];
    for (int i = 0; i < 9; ++i) Ks[i] = src_intr[v*9 + i];
    Ks[0]*=0.5f; Ks[2]*=0.5f; Ks[4]*=0.5f; Ks[5]*=0.5f;
    float KR[9];
    for (int r = 0; r < 3; ++r)
      for (int c = 0; c < 3; ++c)
        KR[r*3+c] = Ks[r*3+0]*T[0*4+c] + Ks[r*3+1]*T[1*4+c] + Ks[r*3+2]*T[2*4+c];
    float A[9];
    for (int r = 0; r < 3; ++r)
      for (int c = 0; c < 3; ++c)
        A[r*3+c] = KR[r*3+0]*Kdi[0*3+c] + KR[r*3+1]*Kdi[1*3+c] + KR[r*3+2]*Kdi[2*3+c];
    for (int i = 0; i < 9; ++i) geom[v*12 + i] = A[i];
    for (int r = 0; r < 3; ++r)
      geom[v*12 + 9 + r] = Ks[r*3+0]*T[3] + Ks[r*3+1]*T[7] + Ks[r*3+2]*T[11];
  }
}

__global__ void k_warp(const float* __restrict__ feats, const float* __restrict__ geom,
                       float depth, float* __restrict__ warped, int H, int W) {
  int idx = blockIdx.x * blockDim.x + threadIdx.x;
  int HW = H * W;
  if (idx >= 3 * HW) return;
  int v = idx / HW, p = idx - v * HW;
  int py = p / W, px = p - py * W;
  const float* g = geom + v * 12;
  float fx = (float)px, fy = (float)py;
  float bx = g[0]*fx + g[1]*fy + g[2];
  float by = g[3]*fx + g[4]*fy + g[5];
  float bz = g[6]*fx + g[7]*fy + g[8];
  float pX = depth * bx + g[9];
  float pY = depth * by + g[10];
  float pZ = depth * bz + g[11];
  float z = (fabsf(pZ) < 1e-6f) ? 1e-6f : pZ;
  float x = pX / z, y = pY / z;
  float x0 = floorf(x), y0 = floorf(y);
  float wx1 = x - x0, wy1 = y - y0;
  const float* f = feats + v * 16 * HW;
  float acc[16];
#pragma unroll
  for (int c = 0; c < 16; ++c) acc[c] = 0.f;
  for (int dy = 0; dy < 2; ++dy) {
    float wy = dy ? wy1 : 1.f - wy1;
    float yi = y0 + (float)dy;
    for (int dx = 0; dx < 2; ++dx) {
      float wx = dx ? wx1 : 1.f - wx1;
      float xi = x0 + (float)dx;
      bool valid = (xi >= 0.f) && (xi <= (float)(W-1)) && (yi >= 0.f) && (yi <= (float)(H-1));
      float wgt = wx * wy * (valid ? 1.f : 0.f);
      int xc = (int)fminf(fmaxf(xi, 0.f), (float)(W-1));
      int yc = (int)fminf(fmaxf(yi, 0.f), (float)(H-1));
      const float* fp = f + yc * W + xc;
#pragma unroll
      for (int c = 0; c < 16; ++c) acc[c] += fp[c * HW] * wgt;
    }
  }
#pragma unroll
  for (int c = 0; c < 16; ++c) warped[(v * 16 + c) * HW + p] = acc[c];
}

__global__ void k_costvc(const float* __restrict__ warped, float* __restrict__ vc, int HW) {
  int p = blockIdx.x * blockDim.x + threadIdx.x;
  if (p >= HW) return;
  float f[3][16];
  for (int v = 0; v < 3; ++v)
    for (int c = 0; c < 16; ++c) f[v][c] = warped[(v * 16 + c) * HW + p];
  float cost[3][3];
  for (int v = 0; v < 3; ++v)
    for (int m = 0; m < 3; ++m) {
      float s = 0.f;
      for (int c = 0; c < 16; ++c) s += f[v][c] * f[m][c];
      cost[v][m] = s;
    }
  float Bm[3], Cm = 0.f;
  for (int m = 0; m < 3; ++m) { Bm[m] = (cost[0][m]+cost[1][m]+cost[2][m]) * (1.f/3.f); Cm += Bm[m]; }
  Cm *= (1.f/3.f);
  for (int v = 0; v < 3; ++v) {
    for (int c = 0; c < 16; ++c) vc[(v*18 + c) * HW + p] = f[v][c];
    vc[(v*18 + 16) * HW + p] = Bm[v];
    vc[(v*18 + 17) * HW + p] = Cm;
  }
}

__global__ void k_copychan(float* __restrict__ dst, const float* __restrict__ src,
                           int B, int C, int HW, int dstC, int dstOff) {
  int i = blockIdx.x * blockDim.x + threadIdx.x;
  int n = B * C * HW;
  if (i >= n) return;
  int b = i / (C * HW), r = i - b * C * HW;
  int c = r / HW, p = r - c * HW;
  dst[(b * dstC + dstOff + c) * HW + p] = src[i];
}

__global__ void k_rh(float* __restrict__ dst, const float* __restrict__ zr,
                     const float* __restrict__ h, int B, int Ch, int HW, int dstC, int dstOff) {
  int i = blockIdx.x * blockDim.x + threadIdx.x;
  int n = B * Ch * HW;
  if (i >= n) return;
  int b = i / (Ch * HW), r = i - b * Ch * HW;
  int c = r / HW, p = r - c * HW;
  float rg = zr[(b * 2 * Ch + Ch + c) * HW + p];
  dst[(b * dstC + dstOff + c) * HW + p] = rg * h[(b * Ch + c) * HW + p];
}

__global__ void k_gruupd(float* __restrict__ h, const float* __restrict__ zr,
                         const float* __restrict__ hc, int B, int Ch, int HW) {
  int i = blockIdx.x * blockDim.x + threadIdx.x;
  int n = B * Ch * HW;
  if (i >= n) return;
  int b = i / (Ch * HW), r = i - b * Ch * HW;
  int c = r / HW, p = r - c * HW;
  float z = zr[(b * 2 * Ch + c) * HW + p];
  h[i] = (1.f - z) * h[i] + z * hc[i];
}

__global__ void k_maxpool(float* __restrict__ out, const float* __restrict__ in,
                          int B, int C, int H, int W) {
  int Ho = H >> 1, Wo = W >> 1;
  int i = blockIdx.x * blockDim.x + threadIdx.x;
  int n = B * C * Ho * Wo;
  if (i >= n) return;
  int bc = i / (Ho * Wo), r = i - bc * Ho * Wo;
  int y = r / Wo, x = r - y * Wo;
  const float* ip = in + (bc * H + 2 * y) * W + 2 * x;
  out[i] = fmaxf(fmaxf(ip[0], ip[1]), fmaxf(ip[W], ip[W + 1]));
}

__global__ void k_deconv(float* __restrict__ y, const float* __restrict__ x,
                         const float* __restrict__ w, const float* __restrict__ bias,
                         int B, int Cin, int Cout, int Hin, int Win, int Hout, int Wout) {
  int i = blockIdx.x * blockDim.x + threadIdx.x;
  int n = B * Cout * Hout * Wout;
  if (i >= n) return;
  int b = i / (Cout * Hout * Wout), r = i - b * Cout * Hout * Wout;
  int co = r / (Hout * Wout); r -= co * Hout * Wout;
  int oy = r / Wout, ox = r - oy * Wout;
  int iy_lo = (oy >= 15) ? ((oy - 14) >> 1) : 0;
  int iy_hi = imin_(Hin - 1, oy >> 1);
  int ix_lo = (ox >= 15) ? ((ox - 14) >> 1) : 0;
  int ix_hi = imin_(Win - 1, ox >> 1);
  float s = bias[co];
  for (int ci = 0; ci < Cin; ++ci) {
    const float* xp = x + (b * Cin + ci) * Hin * Win;
    const float* wp = w + (ci * Cout + co) * 256;
    for (int iy = iy_lo; iy <= iy_hi; ++iy) {
      int ky = 2 * iy - oy + 15;
      for (int ix = ix_lo; ix <= ix_hi; ++ix) {
        int kx = 2 * ix - ox + 15;
        s += xp[iy * Win + ix] * wp[(15 - ky) * 16 + (15 - kx)];
      }
    }
  }
  y[i] = s;
}

__global__ void k_gnstats(const float* __restrict__ y, float* __restrict__ stats, int CHW) {
  int b = blockIdx.x;
  const float* p = y + (size_t)b * CHW;
  __shared__ float ss[256], sq[256];
  float s = 0.f, q = 0.f;
  for (int i = threadIdx.x; i < CHW; i += blockDim.x) { float v = p[i]; s += v; q += v * v; }
  ss[threadIdx.x] = s; sq[threadIdx.x] = q;
  __syncthreads();
  for (int st = 128; st > 0; st >>= 1) {
    if ((int)threadIdx.x < st) { ss[threadIdx.x] += ss[threadIdx.x + st]; sq[threadIdx.x] += sq[threadIdx.x + st]; }
    __syncthreads();
  }
  if (threadIdx.x == 0) {
    float mu = ss[0] / (float)CHW;
    float var = sq[0] / (float)CHW - mu * mu;
    stats[2 * b] = mu;
    stats[2 * b + 1] = rsqrtf(var + 1e-5f);
  }
}

__global__ void k_gnapply(float* __restrict__ y, const float* __restrict__ stats,
                          const float* __restrict__ g, const float* __restrict__ beta,
                          int B, int C, int HW) {
  int i = blockIdx.x * blockDim.x + threadIdx.x;
  int n = B * C * HW;
  if (i >= n) return;
  int b = i / (C * HW), r = i - b * C * HW;
  int c = r / HW;
  y[i] = (y[i] - stats[2*b]) * stats[2*b + 1] * g[c] + beta[c];
}

__global__ void k_cropcopy(float* __restrict__ dst, const float* __restrict__ src,
                           int B, int C, int Hs, int Ws, int Hd, int Wd,
                           int dstC, int dstOff) {
  int i = blockIdx.x * blockDim.x + threadIdx.x;
  int n = B * C * Hd * Wd;
  if (i >= n) return;
  int b = i / (C * Hd * Wd), r = i - b * C * Hd * Wd;
  int c = r / (Hd * Wd); r -= c * (Hd * Wd);
  int y = r / Wd, x = r - y * Wd;
  int t = (Hs - Hd) >> 1, l = (Ws - Wd) >> 1;
  dst[(b * dstC + dstOff + c) * Hd * Wd + y * Wd + x] =
      src[((b * C + c) * Hs + t + y) * Ws + l + x];
}

__global__ void k_split3(const float* __restrict__ c3, float* __restrict__ sw,
                         float* __restrict__ m8, int HW) {
  int p = blockIdx.x * blockDim.x + threadIdx.x;
  if (p >= HW) return;
  for (int v = 0; v < 3; ++v) sw[v * HW + p] = c3[(v * 9 + 0) * HW + p];
  for (int c = 0; c < 8; ++c) {
    float s = c3[(0*9 + 1 + c) * HW + p] + c3[(1*9 + 1 + c) * HW + p] + c3[(2*9 + 1 + c) * HW + p];
    m8[c * HW + p] = s * (1.f / 3.f);
  }
}

__global__ void k_out_dp(float* __restrict__ out, const float* __restrict__ dp_all) {
  int i = blockIdx.x * blockDim.x + threadIdx.x;
  const int n = 32 * 256 * 256;
  if (i >= n) return;
  int d = i / 65536, r = i - d * 65536;
  int Y = r / 256, X = r - Y * 256;
  out[i] = dp_all[d * 16384 + (Y >> 1) * 128 + (X >> 1)];
}

__global__ void k_out_sw(float* __restrict__ out, const float* __restrict__ sw_all) {
  int i = blockIdx.x * blockDim.x + threadIdx.x;
  const int n = 3 * 32 * 256 * 256;
  if (i >= n) return;
  int v = i / (32 * 65536), r = i - v * 32 * 65536;
  int d = r / 65536; r -= d * 65536;
  int Y = r / 256, X = r - Y * 256;
  out[2097152 + i] = sw_all[(d * 3 + v) * 16384 + (Y >> 1) * 128 + (X >> 1)];
}

// ---------------------------------------------------------------------------
struct Layer {
  const float* w; const float* b;
  int Cin, Cout, Cpad, shift, H, W, dil;
  int Hp, Wp, Kround, nchunk;
  _Float16* wpack; int* offtab;
};

static void conv_run(hipStream_t stream, const Layer& L, const float* src,
                     _Float16* pad, const float* gamma, const float* beta,
                     float* out, int B, int act) {
  int np = B * L.Cpad * L.Hp * L.Wp;
  k_padcvt<<<(np + 255) / 256, 256, 0, stream>>>(src, pad, B, L.Cin, L.Cpad, L.H, L.W, L.dil);
  int tiles  = B * ((L.H * L.W + 31) / 32);       // 32 pixels per wave
  int blocks = (tiles + 7) / 8;                   // 8 waves / block
  conv3x3_wmma3<<<blocks, 256, 0, stream>>>(pad, L.wpack, L.offtab, L.Kround, L.nchunk,
      L.b, gamma, beta, out, B, L.Cpad, L.Cout, L.H, L.W, L.Hp, L.Wp, act);
}

#define LNCH(kern, n, ...) kern<<<((n) + 255) / 256, 256, 0, stream>>>(__VA_ARGS__)

extern "C" void kernel_launch(void* const* d_in, const int* in_sizes, int n_in,
                              void* d_out, int out_size, void* d_ws, size_t ws_size,
                              hipStream_t stream) {
  const float* src_images = (const float*)d_in[0];
  const float* dst_intr   = (const float*)d_in[1];
  const float* dst_extr   = (const float*)d_in[2];
  const float* src_intr   = (const float*)d_in[3];
  const float* src_extr   = (const float*)d_in[4];
#define P(i) ((const float*)d_in[(i)])
  const float *w1=P(5),*b1=P(6),*g1=P(7),*bb1=P(8);
  const float *w2=P(9),*b2=P(10),*g2=P(11),*bb2=P(12);
  const float *w3=P(13),*b3=P(14),*g3=P(15),*bb3=P(16);
  const float *w4=P(17),*b4=P(18);
  const float *wzr0=P(19),*bzr0=P(20),*wh0=P(21),*bh0=P(22);
  const float *wzr1=P(23),*bzr1=P(24),*wh1=P(25),*bh1=P(26);
  const float *wzr2=P(27),*bzr2=P(28),*wh2=P(29),*bh2=P(30);
  const float *wzr3=P(31),*bzr3=P(32),*wh3=P(33),*bh3=P(34);
  const float *wzr4=P(35),*bzr4=P(36),*wh4=P(37),*bh4=P(38);
  const float *wd2=P(39),*bd2=P(40),*gd2=P(41),*betad2=P(42);
  const float *wd3=P(43),*bd3=P(44),*gd3=P(45),*betad3=P(46);
  const float *wc3=P(47),*bc3=P(48),*wc4=P(49),*bc4=P(50);
#undef P
  float* outp = (float*)d_out;

  const int H = 128, W = 128, HW = H * W;

  float* ws = (float*)d_ws;
  size_t off = 0;
  auto alloc  = [&](size_t n) { float* p = ws + off; off += (n + 63) & ~(size_t)63; return p; };
  auto allocH = [&](size_t n) { return (_Float16*)alloc((n + 1) / 2); };
  auto allocI = [&](size_t n) { return (int*)alloc(n); };

  float* geom    = alloc(64);
  float* imgs    = alloc(9 * HW);
  float* t1      = alloc(3 * 8 * HW);
  float* t2      = alloc(3 * 8 * HW);
  float* feats   = alloc(3 * 16 * HW);
  float* warped  = alloc(3 * 16 * HW);
  float* vc      = alloc(3 * 18 * HW);
  float* s0      = alloc(3 * 8 * HW);
  float* s1      = alloc(3 * 4 * 4096);
  float* s2      = alloc(3 * 4 * 1024);
  float* s3      = alloc(3 * 4 * 4096);
  float* s4      = alloc(4 * HW);
  float* cat0    = alloc(3 * 26 * HW);
  float* zr0     = alloc(3 * 16 * HW);
  float* hc0     = alloc(3 * 8 * HW);
  float* pool0   = alloc(3 * 8 * 4096);
  float* cat1    = alloc(3 * 12 * 4096);
  float* zr1     = alloc(3 * 8 * 4096);
  float* hc1     = alloc(3 * 4 * 4096);
  float* pool1   = alloc(3 * 4 * 1024);
  float* cat2    = alloc(3 * 8 * 1024);
  float* zr2     = alloc(3 * 8 * 1024);
  float* hc2     = alloc(3 * 4 * 1024);
  float* dec2y   = alloc(3 * 3 * 78 * 78);
  float* stats   = alloc(8);
  float* cat3    = alloc(3 * 7 * 4096);
  float* cat3g   = alloc(3 * 11 * 4096);
  float* zr3     = alloc(3 * 8 * 4096);
  float* hc3     = alloc(3 * 4 * 4096);
  float* dec3y   = alloc(3 * 3 * 142 * 142);
  float* cat4    = alloc(3 * 11 * HW);
  float* c3out   = alloc(3 * 9 * HW);
  float* m8      = alloc(8 * HW);
  float* cat5    = alloc(12 * HW);
  float* zr4     = alloc(8 * HW);
  float* hc4     = alloc(4 * HW);
  float* c4out   = alloc(HW);
  float* sw_all  = alloc(32 * 3 * HW);
  float* dp_all  = alloc(32 * HW);

  _Float16* ph_imgs = allocH(3 * 4  * 130 * 130);
  _Float16* ph_a    = allocH(3 * 8  * 130 * 130);
  _Float16* ph_b    = allocH(3 * 8  * 132 * 132);
  _Float16* ph_cat0 = allocH(3 * 32 * 130 * 130);
  _Float16* ph_cat1 = allocH(3 * 16 * 66  * 66);
  _Float16* ph_cat2 = allocH(3 * 8  * 34  * 34);
  _Float16* ph_cat3 = allocH(3 * 16 * 66  * 66);
  _Float16* ph_cat4 = allocH(3 * 16 * 130 * 130);
  _Float16* ph_cat5 = allocH(1 * 16 * 130 * 130);
  _Float16* ph_s4   = allocH(1 * 4  * 130 * 130);
  (void)ws_size; (void)n_in; (void)in_sizes; (void)out_size;

  Layer L[16] = {
    { w1,   b1,   3,  8,  4, 2, 128, 128, 1 },
    { w2,   b2,   8,  8,  8, 3, 128, 128, 1 },
    { w3,   b3,   8,  8,  8, 3, 128, 128, 2 },
    { w4,   b4,   8, 16,  8, 3, 128, 128, 1 },
    { wzr0, bzr0, 26, 16, 32, 5, 128, 128, 1 },
    { wh0,  bh0,  26,  8, 32, 5, 128, 128, 1 },
    { wzr1, bzr1, 12,  8, 16, 4,  64,  64, 1 },
    { wh1,  bh1,  12,  4, 16, 4,  64,  64, 1 },
    { wzr2, bzr2,  8,  8,  8, 3,  32,  32, 1 },
    { wh2,  bh2,   8,  4,  8, 3,  32,  32, 1 },
    { wzr3, bzr3, 11,  8, 16, 4,  64,  64, 1 },
    { wh3,  bh3,  11,  4, 16, 4,  64,  64, 1 },
    { wzr4, bzr4, 12,  8, 16, 4, 128, 128, 1 },
    { wh4,  bh4,  12,  4, 16, 4, 128, 128, 1 },
    { wc3,  bc3,  11,  9, 16, 4, 128, 128, 1 },
    { wc4,  bc4,   4,  1,  4, 2, 128, 128, 1 },
  };
  for (int i = 0; i < 16; ++i) {
    L[i].Hp = L[i].H + 2 * L[i].dil;
    L[i].Wp = L[i].W + 2 * L[i].dil;
    L[i].nchunk = (9 * L[i].Cpad + 31) / 32;
    L[i].Kround = L[i].nchunk * 32;
    L[i].wpack  = allocH((size_t)L[i].nchunk * 512);
    L[i].offtab = allocI(L[i].Kround);
    int cmask = L[i].Cpad - 1;
    LNCH(k_packw, L[i].nchunk * 512, L[i].w, L[i].wpack, L[i].Cin, L[i].Cout, cmask, L[i].shift, L[i].Kround);
    LNCH(k_offtab, L[i].Kround, L[i].offtab, cmask, L[i].shift, L[i].Hp, L[i].Wp, L[i].dil, L[i].Kround);
  }

  hipMemsetAsync(s0, 0, 3 * 8 * HW * sizeof(float), stream);
  hipMemsetAsync(s1, 0, 3 * 4 * 4096 * sizeof(float), stream);
  hipMemsetAsync(s2, 0, 3 * 4 * 1024 * sizeof(float), stream);
  hipMemsetAsync(s3, 0, 3 * 4 * 4096 * sizeof(float), stream);
  hipMemsetAsync(s4, 0, 4 * HW * sizeof(float), stream);

  LNCH(k_downsample, 9 * HW, src_images, imgs);
  k_setup_geom<<<1, 32, 0, stream>>>(dst_intr, dst_extr, src_intr, src_extr, geom);
  conv_run(stream, L[0], imgs, ph_imgs, g1, bb1, t1, 3, 1);
  conv_run(stream, L[1], t1, ph_a, g2, bb2, t2, 3, 1);
  conv_run(stream, L[2], t2, ph_b, g3, bb3, t1, 3, 1);
  conv_run(stream, L[3], t1, ph_a, nullptr, nullptr, feats, 3, 0);

  for (int step = 0; step < 32; ++step) {
    float depth = 0.5f + (9.5f / 31.f) * (float)step;
    LNCH(k_warp, 3 * HW, feats, geom, depth, warped, H, W);
    LNCH(k_costvc, HW, warped, vc, HW);

    // cell0: gru(18, 8) @128
    LNCH(k_copychan, 3 * 18 * HW, cat0, vc, 3, 18, HW, 26, 0);
    LNCH(k_copychan, 3 * 8 * HW,  cat0, s0, 3, 8,  HW, 26, 18);
    conv_run(stream, L[4], cat0, ph_cat0, nullptr, nullptr, zr0, 3, 2);
    LNCH(k_rh, 3 * 8 * HW, cat0, zr0, s0, 3, 8, HW, 26, 18);
    conv_run(stream, L[5], cat0, ph_cat0, nullptr, nullptr, hc0, 3, 3);
    LNCH(k_gruupd, 3 * 8 * HW, s0, zr0, hc0, 3, 8, HW);

    // cell1: gru(8, 4) @64
    LNCH(k_maxpool, 3 * 8 * 4096, pool0, s0, 3, 8, 128, 128);
    LNCH(k_copychan, 3 * 8 * 4096, cat1, pool0, 3, 8, 4096, 12, 0);
    LNCH(k_copychan, 3 * 4 * 4096, cat1, s1,    3, 4, 4096, 12, 8);
    conv_run(stream, L[6], cat1, ph_cat1, nullptr, nullptr, zr1, 3, 2);
    LNCH(k_rh, 3 * 4 * 4096, cat1, zr1, s1, 3, 4, 4096, 12, 8);
    conv_run(stream, L[7], cat1, ph_cat1, nullptr, nullptr, hc1, 3, 3);
    LNCH(k_gruupd, 3 * 4 * 4096, s1, zr1, hc1, 3, 4, 4096);

    // cell2: gru(4, 4) @32
    LNCH(k_maxpool, 3 * 4 * 1024, pool1, s1, 3, 4, 64, 64);
    LNCH(k_copychan, 3 * 4 * 1024, cat2, pool1, 3, 4, 1024, 8, 0);
    LNCH(k_copychan, 3 * 4 * 1024, cat2, s2,    3, 4, 1024, 8, 4);
    conv_run(stream, L[8], cat2, ph_cat2, nullptr, nullptr, zr2, 3, 2);
    LNCH(k_rh, 3 * 4 * 1024, cat2, zr2, s2, 3, 4, 1024, 8, 4);
    conv_run(stream, L[9], cat2, ph_cat2, nullptr, nullptr, hc2, 3, 3);
    LNCH(k_gruupd, 3 * 4 * 1024, s2, zr2, hc2, 3, 4, 1024);

    // deconv2 + GN, crop to 64, concat with s1
    LNCH(k_deconv, 3 * 3 * 78 * 78, dec2y, s2, wd2, bd2, 3, 4, 3, 32, 32, 78, 78);
    k_gnstats<<<3, 256, 0, stream>>>(dec2y, stats, 3 * 78 * 78);
    LNCH(k_gnapply, 3 * 3 * 78 * 78, dec2y, stats, gd2, betad2, 3, 3, 78 * 78);
    LNCH(k_cropcopy, 3 * 3 * 4096, cat3, dec2y, 3, 3, 78, 78, 64, 64, 7, 0);
    LNCH(k_copychan, 3 * 4 * 4096, cat3, s1, 3, 4, 4096, 7, 3);

    // cell3: gru(7, 4) @64
    LNCH(k_copychan, 3 * 7 * 4096, cat3g, cat3, 3, 7, 4096, 11, 0);
    LNCH(k_copychan, 3 * 4 * 4096, cat3g, s3,   3, 4, 4096, 11, 7);
    conv_run(stream, L[10], cat3g, ph_cat3, nullptr, nullptr, zr3, 3, 2);
    LNCH(k_rh, 3 * 4 * 4096, cat3g, zr3, s3, 3, 4, 4096, 11, 7);
    conv_run(stream, L[11], cat3g, ph_cat3, nullptr, nullptr, hc3, 3, 3);
    LNCH(k_gruupd, 3 * 4 * 4096, s3, zr3, hc3, 3, 4, 4096);

    // deconv3 + GN, crop to 128, concat with s0
    LNCH(k_deconv, 3 * 3 * 142 * 142, dec3y, s3, wd3, bd3, 3, 4, 3, 64, 64, 142, 142);
    k_gnstats<<<3, 256, 0, stream>>>(dec3y, stats, 3 * 142 * 142);
    LNCH(k_gnapply, 3 * 3 * 142 * 142, dec3y, stats, gd3, betad3, 3, 3, 142 * 142);
    LNCH(k_cropcopy, 3 * 3 * HW, cat4, dec3y, 3, 3, 142, 142, 128, 128, 11, 0);
    LNCH(k_copychan, 3 * 8 * HW, cat4, s0, 3, 8, HW, 11, 3);

    // conv3 head -> per-view weight + mean(1:9) for cell4
    conv_run(stream, L[14], cat4, ph_cat4, nullptr, nullptr, c3out, 3, 0);
    LNCH(k_split3, HW, c3out, sw_all + (size_t)step * 3 * HW, m8, HW);

    // cell4: gru(8, 4) @128, B=1
    LNCH(k_copychan, 8 * HW, cat5, m8, 1, 8, HW, 12, 0);
    LNCH(k_copychan, 4 * HW, cat5, s4, 1, 4, HW, 12, 8);
    conv_run(stream, L[12], cat5, ph_cat5, nullptr, nullptr, zr4, 1, 2);
    LNCH(k_rh, 4 * HW, cat5, zr4, s4, 1, 4, HW, 12, 8);
    conv_run(stream, L[13], cat5, ph_cat5, nullptr, nullptr, hc4, 1, 3);
    LNCH(k_gruupd, 4 * HW, s4, zr4, hc4, 1, 4, HW);
    conv_run(stream, L[15], s4, ph_s4, nullptr, nullptr, c4out, 1, 0);
    LNCH(k_copychan, HW, dp_all + (size_t)step * HW, c4out, 1, 1, HW, 1, 0);
  }

  LNCH(k_out_dp, 32 * 256 * 256, outp, dp_all);
  LNCH(k_out_sw, 3 * 32 * 256 * 256, outp, sw_all);
}
